// TtShiftedWindowAttention_42915313222139
// MI455X (gfx1250) — compile-verified, matrix operand order checked by reading, not run on hardware
//
#include <hip/hip_runtime.h>
#include <hip/hip_bf16.h>

// ---- CDNA5 wave32 WMMA types ----
typedef __attribute__((ext_vector_type(16))) _Float16 v16h;
typedef __attribute__((ext_vector_type(8)))  float    v8f;
typedef __attribute__((ext_vector_type(8)))  _Float16 h8;
typedef __attribute__((ext_vector_type(4)))  _Float16 h4;

#define NHEADS 3
#define CDIM   96      // channels
#define DHEAD  32      // head dim
#define LWIN   64      // window tokens (8x8)
#define HDIM   128     // image H=W
#define SPSTR  72      // padded f16 stride for score/prob buffer
#define VTSTR  72      // padded f16 stride for transposed V

// Load a 16x32 f16 A-fragment (or B^T-fragment, identical layout) for
// v_wmma_f32_16x16x32_f16.  Per ISA 7.12.2: lane L holds row (L&15);
// lanes 0-15 hold K=0..7,16..23, lanes 16-31 hold K=8..15,24..31.
// Two 16-byte loads per lane (stride must be a multiple of 8 halves).
__device__ __forceinline__ v16h load_frag(const _Float16* base, int stride) {
  const int lane = threadIdx.x & 31;
  const int row  = lane & 15;
  const int kb   = (lane >> 4) << 3;           // 0 or 8 halves
  const _Float16* p = base + row * stride + kb;
  h8 lo = *(const h8*)(p);
  h8 hi = *(const h8*)(p + 16);
  v16h r;
#pragma unroll
  for (int i = 0; i < 8; ++i) { r[i] = lo[i]; r[i + 8] = hi[i]; }
  return r;
}

// One 16x16 output tile of A(MxK) @ B^T where both operands are row-major
// with the given strides; K = 32*ksteps.
template <int KSTEPS>
__device__ __forceinline__ v8f wmma_tile(const _Float16* A, int lda,
                                         const _Float16* B, int ldb) {
  v8f acc = {};
#pragma unroll
  for (int kt = 0; kt < KSTEPS; ++kt) {
    v16h a = load_frag(A + kt * 32, lda);
    v16h b = load_frag(B + kt * 32, ldb);
    acc = __builtin_amdgcn_wmma_f32_16x16x32_f16(false, a, false, b,
                                                 (short)0, acc, false, false);
  }
  return acc;
}

// Swin shifted-window region bucket (pad_H = 128, WS=8, SS=4):
// [0,120) -> 0, [120,124) -> 1, [124,128) -> 2
__device__ __forceinline__ int bucket(int g) {
  return (g < HDIM - 8) ? 0 : ((g < HDIM - 4) ? 1 : 2);
}

// ------------------------------------------------------------------
// Pre-kernel: convert qkv_w (288x96) and proj_w (96x96) f32 -> f16
// ------------------------------------------------------------------
__global__ void cvt_weights(const float* __restrict__ qkv_w,
                            const float* __restrict__ proj_w,
                            _Float16* __restrict__ qkv_w16,
                            _Float16* __restrict__ proj_w16) {
  int t = blockIdx.x * blockDim.x + threadIdx.x;
  if (t < 3 * CDIM * CDIM) qkv_w16[t] = (_Float16)qkv_w[t];
  if (t < CDIM * CDIM)     proj_w16[t] = (_Float16)proj_w[t];
}

// ------------------------------------------------------------------
// Fused shifted-window attention: one block (8 waves) per window.
// ------------------------------------------------------------------
__global__ __launch_bounds__(256)
void swin_attn(const float* __restrict__ x,
               const float* __restrict__ qkv_b,
               const float* __restrict__ proj_b,
               const float* __restrict__ rel_bias,
               const _Float16* __restrict__ qkv_w16,
               const _Float16* __restrict__ proj_w16,
               float* __restrict__ out) {
  __shared__ _Float16 sh_x [LWIN * CDIM];    // input window; reused as attn-out
  __shared__ _Float16 sh_q [LWIN * CDIM];    // pre-scaled q (all heads)
  __shared__ _Float16 sh_k [LWIN * CDIM];
  __shared__ _Float16 sh_vT[CDIM * VTSTR];   // V transposed: [feature][token]
  __shared__ _Float16 sh_SP[LWIN * SPSTR];   // scores -> probs, f16 in place

  const int tid  = threadIdx.x;
  const int lane = tid & 31;
  const int wave = tid >> 5;
  const int blk  = blockIdx.x;
  const int b    = blk >> 8;        // batch
  const int wy   = (blk >> 4) & 15; // window row
  const int wx   = blk & 15;        // window col

  // warm L2/WGP$ with the (L2-resident) f16 weight tables: global_prefetch_b8
  __builtin_prefetch(qkv_w16 + tid * 108, 0, 3);   // 27648 halves / 256 thr
  __builtin_prefetch(proj_w16 + tid * 36, 0, 3);   //  9216 halves / 256 thr

  // ---- 1) gather shifted window (roll -4,-4) into LDS as f16 ----
  {
    // 64*96 f32 = 1536 float4 chunks, 6 per thread
#pragma unroll
    for (int e = 0; e < 6; ++e) {
      int idx = tid * 6 + e;
      int i   = idx / (CDIM / 4);        // token 0..63
      int c4  = (idx % (CDIM / 4)) * 4;  // channel base
      int r = i >> 3, c = i & 7;
      int sy = (wy * 8 + r + 4) & (HDIM - 1);
      int sx = (wx * 8 + c + 4) & (HDIM - 1);
      const float4 v = *(const float4*)(x + (((size_t)b * HDIM + sy) * HDIM + sx) * CDIM + c4);
      h4 hv; hv[0] = (_Float16)v.x; hv[1] = (_Float16)v.y;
             hv[2] = (_Float16)v.z; hv[3] = (_Float16)v.w;
      *(h4*)(&sh_x[i * CDIM + c4]) = hv;
    }
  }
  __syncthreads();

  // ---- 2) qkv = xw @ qkv_w^T + b : three branch-free tile loops ----
  // q tiles: output features [0,96)
  for (int t = wave; t < 24; t += 8) {
    int mt = t & 3, nt = t >> 2;
    v8f acc = wmma_tile<3>(sh_x + mt * 16 * CDIM, CDIM,
                           qkv_w16 + (nt * 16) * CDIM, CDIM);
    int n = nt * 16 + (lane & 15);
    float bias = qkv_b[n];
#pragma unroll
    for (int rr = 0; rr < 8; ++rr) {
      int qp = mt * 16 + rr + ((lane >> 4) << 3);
      sh_q[qp * CDIM + n] = (_Float16)((acc[rr] + bias) * 0.17677669529663687f);
    }
  }
  // k tiles: output features [96,192)
  for (int t = wave; t < 24; t += 8) {
    int mt = t & 3, nt = (t >> 2) + 6;
    v8f acc = wmma_tile<3>(sh_x + mt * 16 * CDIM, CDIM,
                           qkv_w16 + (nt * 16) * CDIM, CDIM);
    int n = nt * 16 + (lane & 15);
    float bias = qkv_b[n];
    int cch = n - CDIM;
#pragma unroll
    for (int rr = 0; rr < 8; ++rr) {
      int qp = mt * 16 + rr + ((lane >> 4) << 3);
      sh_k[qp * CDIM + cch] = (_Float16)(acc[rr] + bias);
    }
  }
  // v tiles: output features [192,288), stored transposed
  for (int t = wave; t < 24; t += 8) {
    int mt = t & 3, nt = (t >> 2) + 12;
    v8f acc = wmma_tile<3>(sh_x + mt * 16 * CDIM, CDIM,
                           qkv_w16 + (nt * 16) * CDIM, CDIM);
    int n = nt * 16 + (lane & 15);
    float bias = qkv_b[n];
    int cch = n - 2 * CDIM;
#pragma unroll
    for (int rr = 0; rr < 8; ++rr) {
      int qp = mt * 16 + rr + ((lane >> 4) << 3);
      sh_vT[cch * VTSTR + qp] = (_Float16)(acc[rr] + bias);
    }
  }
  __syncthreads();

  // ---- 3) per-head attention ----
  for (int h = 0; h < NHEADS; ++h) {
    // 3a) S = q_h @ k_h^T (+rel_bias +shift mask) : 4x4 tiles, K=32
    for (int t = wave; t < 16; t += 8) {
      int mt = t & 3, nt = t >> 2;
      v8f acc = wmma_tile<1>(sh_q + mt * 16 * CDIM + h * DHEAD, CDIM,
                             sh_k + nt * 16 * CDIM + h * DHEAD, CDIM);
      int kp  = nt * 16 + (lane & 15);
      int idk = 3 * bucket(wy * 8 + (kp >> 3)) + bucket(wx * 8 + (kp & 7));
#pragma unroll
      for (int rr = 0; rr < 8; ++rr) {
        int qp  = mt * 16 + rr + ((lane >> 4) << 3);
        int idq = 3 * bucket(wy * 8 + (qp >> 3)) + bucket(wx * 8 + (qp & 7));
        float val = acc[rr] + rel_bias[(h * LWIN + qp) * LWIN + kp];
        if (idq != idk) val -= 100.0f;
        sh_SP[qp * SPSTR + kp] = (_Float16)val;
      }
    }
    __syncthreads();

    // 3b) row softmax in place (f16), threads 0..63 each own a row
    if (tid < LWIN) {
      float mx = -3.0e38f;
#pragma unroll 8
      for (int j = 0; j < LWIN; ++j) mx = fmaxf(mx, (float)sh_SP[tid * SPSTR + j]);
      float sum = 0.f;
#pragma unroll 8
      for (int j = 0; j < LWIN; ++j) {
        float e = __expf((float)sh_SP[tid * SPSTR + j] - mx);
        sum += e;
        sh_SP[tid * SPSTR + j] = (_Float16)e;
      }
      float inv = 1.0f / sum;
#pragma unroll 8
      for (int j = 0; j < LWIN; ++j)
        sh_SP[tid * SPSTR + j] = (_Float16)((float)sh_SP[tid * SPSTR + j] * inv);
    }
    __syncthreads();

    // 3c) O_h = P @ V_h : 4x2 tiles (one per wave), K=64 (2 steps)
    {
      int mt = wave & 3, nt = wave >> 2;
      v8f acc = wmma_tile<2>(sh_SP + mt * 16 * SPSTR, SPSTR,
                             sh_vT + (h * DHEAD + nt * 16) * VTSTR, VTSTR);
      int n = h * DHEAD + nt * 16 + (lane & 15);
#pragma unroll
      for (int rr = 0; rr < 8; ++rr) {
        int qp = mt * 16 + rr + ((lane >> 4) << 3);
        sh_x[qp * CDIM + n] = (_Float16)acc[rr];   // reuse sh_x as attn-out
      }
    }
    __syncthreads();
  }

  // ---- 4) proj + bias, fused un-shift scatter : 4x6 tiles, K=3 ----
  for (int t = wave; t < 24; t += 8) {
    int mt = t & 3, nt = t >> 2;
    v8f acc = wmma_tile<3>(sh_x + mt * 16 * CDIM, CDIM,
                           proj_w16 + (nt * 16) * CDIM, CDIM);
    int f = nt * 16 + (lane & 15);
    float bias = proj_b[f];
#pragma unroll
    for (int rr = 0; rr < 8; ++rr) {
      int qp = mt * 16 + rr + ((lane >> 4) << 3);
      int oy = (wy * 8 + (qp >> 3) + 4) & (HDIM - 1);  // roll back +4
      int ox = (wx * 8 + (qp & 7) + 4) & (HDIM - 1);
      out[(((size_t)b * HDIM + oy) * HDIM + ox) * CDIM + f] = acc[rr] + bias;
    }
  }
}

extern "C" void kernel_launch(void* const* d_in, const int* in_sizes, int n_in,
                              void* d_out, int out_size, void* d_ws, size_t ws_size,
                              hipStream_t stream) {
  const float* x      = (const float*)d_in[0];
  const float* qkv_w  = (const float*)d_in[1];
  const float* qkv_b  = (const float*)d_in[2];
  const float* proj_w = (const float*)d_in[3];
  const float* proj_b = (const float*)d_in[4];
  const float* relb   = (const float*)d_in[5];
  float* out = (float*)d_out;

  _Float16* qkv_w16  = (_Float16*)d_ws;                 // 288*96 f16 = 55296 B
  _Float16* proj_w16 = qkv_w16 + 3 * CDIM * CDIM;       // 96*96  f16 = 18432 B

  cvt_weights<<<(3 * CDIM * CDIM + 255) / 256, 256, 0, stream>>>(
      qkv_w, proj_w, qkv_w16, proj_w16);

  // 16 batches * 16*16 windows = 4096 blocks, 256 threads (8 wave32)
  swin_attn<<<16 * 16 * 16, 256, 0, stream>>>(
      x, qkv_b, proj_b, relb, qkv_w16, proj_w16, out);
}
